// AttentionNorm_17798344474640
// MI455X (gfx1250) — matrix operand order repaired and checked by reading, not compiled, not run on hardware
//
#include <hip/hip_runtime.h>

typedef __attribute__((ext_vector_type(16))) _Float16 v16h;
typedef __attribute__((ext_vector_type(8)))  float    v8f;
typedef _Float16 h16;

#define CHN   128
#define C2D   64
#define HWD   96
#define NPIX  9216      // 96*96
#define MPIX  2304      // 48*48
#define NB    4
#define HIN   192
#define TEMP_F 5.0f

union Frag16 { v16h h; unsigned u[8]; };
union Frag32 { v8f f; float x[8]; };

__device__ __forceinline__ unsigned pack2h(float a, float b) {
  union { h16 h[2]; unsigned u; } t;
  t.h[0] = (h16)a; t.h[1] = (h16)b;
  return t.u;
}

// A-matrix 16x32 f16 layout (ISA 7.12.2): lane holds row (l&15); reg v holds K-pair
//   k = 2v + (v>=4?8:0) + (hi?8:0)
// B-matrix 32x16 f16 layout: lane holds col (l&15); reg v holds K-pair
//   k = 2v + (v>=4?8:0) + (hi?8:0)  (same order as A).
template<bool IS_A>
__device__ __forceinline__ void load_frag(Frag16& f, const h16* base, int row, int ld,
                                          int k0, int hi) {
#pragma unroll
  for (int v = 0; v < 8; ++v) {
    int k = k0 + (IS_A ? (2*v + (v >= 4 ? 8 : 0) + (hi ? 8 : 0))
                       : (2*v + (v >= 4 ? 8 : 0) + (hi ? 8 : 0)));
    f.u[v] = *(const unsigned*)(base + (size_t)row * ld + k);
  }
}

__device__ __forceinline__ void wmma_f16(Frag32& c, const Frag16& a, const Frag16& b) {
  c.f = __builtin_amdgcn_wmma_f32_16x16x32_f16(false, a.h, false, b.h,
                                               (short)0, c.f, false, false);
}

__device__ __forceinline__ void wave_lds_sync() {
  asm volatile("s_wait_dscnt 0" ::: "memory");
  __builtin_amdgcn_wave_barrier();
}

// ---------------------------------------------------------------- weight prep
__global__ void prep_convert_kernel(const float* d0w, const float* d1w,
                                    const float* tw, const float* pw, const float* gw,
                                    h16* wh0, h16* wh1, h16* thw, h16* phw, h16* ghw) {
  const int S0 = CHN * CHN * 4;   // 65536
  const int S1 = C2D * CHN;       // 8192
  const int total = 2 * S0 + 3 * S1;
  int stride = gridDim.x * blockDim.x;
  for (int t = blockIdx.x * blockDim.x + threadIdx.x; t < total; t += stride) {
    if      (t < S0)            wh0[t]                = (h16)d0w[t];
    else if (t < 2*S0)          wh1[t - S0]           = (h16)d1w[t - S0];
    else if (t < 2*S0 + S1)     thw[t - 2*S0]         = (h16)tw[t - 2*S0];
    else if (t < 2*S0 + 2*S1)   phw[t - 2*S0 - S1]    = (h16)pw[t - 2*S0 - S1];
    else                        ghw[t - 2*S0 - 2*S1]  = (h16)gw[t - 2*S0 - 2*S1];
  }
}

// Mf[col=(oc*4+p*2+q)][c2] = sum_ic up_w[ic,oc,p,q] * o_w[ic,c2]
__global__ void prep_mfused_kernel(const float* up_w, const float* o_w, h16* mf) {
  int t = blockIdx.x * blockDim.x + threadIdx.x;
  if (t >= 512 * C2D) return;
  int col = t >> 6, c2 = t & 63;
  float s = 0.f;
  for (int ic = 0; ic < CHN; ++ic)
    s += up_w[ic * 512 + col] * o_w[ic * C2D + c2];
  mf[t] = (h16)s;
}

// ------------------------------------------------- down conv (implicit GEMM K=512)
// out[b,n,oc] = sum_{c,p,q} in[b,c,2i+p,2j+q] * w[oc,c,p,q] + bias[oc]
__global__ __launch_bounds__(256) void down_kernel(
    const float* __restrict__ inx, const float* __restrict__ iny,
    const h16* __restrict__ wh0, const h16* __restrict__ wh1,
    const float* __restrict__ b0, const float* __restrict__ b1,
    h16* __restrict__ xdh, h16* __restrict__ ydh) {
  const int lane = threadIdx.x & 31;
  const int hi = lane >> 4, ln = lane & 15;
  const int wv = threadIdx.x >> 5;
  const int tile = blockIdx.x * 8 + wv;            // 0..2303
  const int b = tile / 576;
  const int n0 = (tile % 576) * 16;
  const bool isY = (blockIdx.y != 0);
  const float* inp = isY ? iny : inx;
  const h16* wh = isY ? wh1 : wh0;
  const float* bias = isY ? b1 : b0;
  h16* outp = (isY ? ydh : xdh) + (size_t)b * NPIX * CHN;

  Frag32 acc[8];
#pragma unroll
  for (int t = 0; t < 8; ++t)
#pragma unroll
    for (int r = 0; r < 8; ++r) acc[t].x[r] = 0.f;

  const int n = n0 + ln;
  const int i = n / HWD, j = n % HWD;
  const float* inb = inp + (size_t)b * CHN * HIN * HIN;

  for (int ks = 0; ks < 16; ++ks) {
    Frag16 a;                                      // activations: fp32 -> f16 on load
#pragma unroll
    for (int v = 0; v < 8; ++v) {
      int k = ks * 32 + 2*v + (v >= 4 ? 8 : 0) + (hi ? 8 : 0);  // k = c*4+p*2+q (q=0)
      int c = k >> 2, p = (k >> 1) & 1;
      const float2 d = *(const float2*)(inb + ((size_t)c * HIN + (2*i + p)) * HIN + 2*j);
      a.u[v] = pack2h(d.x, d.y);
    }
    // batch all weight fragments first so loads clause & overlap with WMMA issue
    Frag16 bw[8];
#pragma unroll
    for (int t = 0; t < 8; ++t)
      load_frag<false>(bw[t], wh, t*16 + ln, CHN*4, ks*32, hi);
#pragma unroll
    for (int t = 0; t < 8; ++t)
      wmma_f16(acc[t], a, bw[t]);
  }
#pragma unroll
  for (int t = 0; t < 8; ++t) {
    const int oc = t*16 + ln;
    const float bs = bias[oc];
#pragma unroll
    for (int r = 0; r < 8; ++r) {
      int rr = n0 + r + hi*8;
      outp[(size_t)rr * CHN + oc] = (h16)(acc[t].x[r] + bs);
    }
  }
}

// --------------------------------------------- 1x1 projections (K=128 GEMMs)
// x-path: theta (scaled by TEMP) + phi_full ; y-path: g_full
__global__ __launch_bounds__(256) void proj_kernel(
    const h16* __restrict__ xdh, const h16* __restrict__ ydh,
    const h16* __restrict__ thw, const h16* __restrict__ phw, const h16* __restrict__ ghw,
    h16* __restrict__ theta, h16* __restrict__ phifull, h16* __restrict__ gfull) {
  const int lane = threadIdx.x & 31;
  const int hi = lane >> 4, ln = lane & 15;
  const int wv = threadIdx.x >> 5;
  const int tile = blockIdx.x * 8 + wv;
  const int b = tile / 576;
  const int n0 = (tile % 576) * 16;
  const bool isY = (blockIdx.y != 0);
  const h16* src = (isY ? ydh : xdh) + (size_t)b * NPIX * CHN;

  Frag16 a[4];
#pragma unroll
  for (int kk = 0; kk < 4; ++kk)
    load_frag<true>(a[kk], src, n0 + ln, CHN, kk*32, hi);

  const int NT = isY ? 4 : 8;
  Frag32 acc[8];
  for (int t = 0; t < 8; ++t) {
    if (t >= NT) break;                            // wave-uniform
#pragma unroll
    for (int r = 0; r < 8; ++r) acc[t].x[r] = 0.f;
    const h16* wm = isY ? ghw : (t < 4 ? thw : phw);
    int tc = isY ? t : (t & 3);
    Frag16 bw[4];
#pragma unroll
    for (int kk = 0; kk < 4; ++kk)
      load_frag<false>(bw[kk], wm, tc*16 + ln, CHN, kk*32, hi);
#pragma unroll
    for (int kk = 0; kk < 4; ++kk)
      wmma_f16(acc[t], a[kk], bw[kk]);
  }
  h16* th_b = theta   + (size_t)b * NPIX * C2D;
  h16* pf_b = phifull + (size_t)b * NPIX * C2D;
  h16* gf_b = gfull   + (size_t)b * NPIX * C2D;
  for (int t = 0; t < 8; ++t) {
    if (t >= NT) break;
    int c = (isY ? t : (t & 3)) * 16 + ln;
#pragma unroll
    for (int r = 0; r < 8; ++r) {
      int rr = n0 + r + hi*8;
      float v = acc[t].x[r];
      if (isY)          gf_b[(size_t)rr * C2D + c] = (h16)v;
      else if (t < 4)   th_b[(size_t)rr * C2D + c] = (h16)(v * TEMP_F); // fold TEMP
      else              pf_b[(size_t)rr * C2D + c] = (h16)v;
    }
  }
}

// ------------------------------------------------ 2x2 max pool (keys + values^T)
__global__ void pool_kernel(const h16* __restrict__ phifull, const h16* __restrict__ gfull,
                            h16* __restrict__ phi, h16* __restrict__ gt) {
  int t = blockIdx.x * blockDim.x + threadIdx.x;
  if (t >= NB * MPIX * C2D) return;
  int c = t & 63;
  int m = (t >> 6) % MPIX;
  int b = t / (C2D * MPIX);
  int ii = m / 48, jj = m % 48;
  int nbase = (2*ii) * HWD + 2*jj;
  const h16* pf = phifull + (size_t)b * NPIX * C2D;
  const h16* gf = gfull   + (size_t)b * NPIX * C2D;
  float p0 = fmaxf(fmaxf((float)pf[(size_t)nbase*C2D + c],       (float)pf[(size_t)(nbase+1)*C2D + c]),
                   fmaxf((float)pf[(size_t)(nbase+HWD)*C2D + c], (float)pf[(size_t)(nbase+HWD+1)*C2D + c]));
  float g0 = fmaxf(fmaxf((float)gf[(size_t)nbase*C2D + c],       (float)gf[(size_t)(nbase+1)*C2D + c]),
                   fmaxf((float)gf[(size_t)(nbase+HWD)*C2D + c], (float)gf[(size_t)(nbase+HWD+1)*C2D + c]));
  phi[((size_t)b * MPIX + m) * C2D + c] = (h16)p0;
  gt [((size_t)b * C2D + c) * MPIX + m] = (h16)g0;   // transposed for PV B-fragments
}

// -------------------------------------------- flash attention (16 queries / wave)
__global__ __launch_bounds__(128) void attn_kernel(
    const h16* __restrict__ theta, const h16* __restrict__ phi,
    const h16* __restrict__ gt, h16* __restrict__ omap) {
  __shared__ h16 lds_p[4][16 * 64];                // per-wave P staging
  const int lane = threadIdx.x & 31;
  const int hi = lane >> 4, ln = lane & 15;
  const int wv = threadIdx.x >> 5;
  const int tile = blockIdx.x * 4 + wv;            // 0..2303
  const int b = tile / 576;
  const int n0 = (tile % 576) * 16;

  const h16* th_b = theta + (size_t)b * NPIX * C2D;
  const h16* ph_b = phi   + (size_t)b * MPIX * C2D;
  const h16* gt_b = gt    + (size_t)b * C2D * MPIX;
  h16* p_l = &lds_p[wv][0];

  Frag16 aq[2];
  load_frag<true>(aq[0], th_b, n0 + ln, C2D, 0, hi);
  load_frag<true>(aq[1], th_b, n0 + ln, C2D, 32, hi);

  Frag32 o[4];
#pragma unroll
  for (int t = 0; t < 4; ++t)
#pragma unroll
    for (int r = 0; r < 8; ++r) o[t].x[r] = 0.f;
  float mrow[8], lrow[8];
#pragma unroll
  for (int r = 0; r < 8; ++r) { mrow[r] = -3.0e38f; lrow[r] = 0.f; }

  for (int m0 = 0; m0 < MPIX; m0 += 64) {
    // S = (TEMP*Q) K^T for 64 keys
    Frag32 s[4];
#pragma unroll
    for (int kt = 0; kt < 4; ++kt) {
#pragma unroll
      for (int r = 0; r < 8; ++r) s[kt].x[r] = 0.f;
      Frag16 bk[2];
      load_frag<false>(bk[0], ph_b, m0 + kt*16 + ln, C2D, 0,  hi);
      load_frag<false>(bk[1], ph_b, m0 + kt*16 + ln, C2D, 32, hi);
      wmma_f16(s[kt], aq[0], bk[0]);
      wmma_f16(s[kt], aq[1], bk[1]);
    }
    // issue value B-fragment loads now; softmax VALU below hides their latency
    Frag16 bv[8];
#pragma unroll
    for (int t = 0; t < 4; ++t) {
      load_frag<false>(bv[t*2+0], gt_b, t*16 + ln, MPIX, m0 +  0, hi);
      load_frag<false>(bv[t*2+1], gt_b, t*16 + ln, MPIX, m0 + 32, hi);
    }
    // online softmax stats (row reductions across the 16-lane half)
    float mnew[8], alpha[8];
#pragma unroll
    for (int r = 0; r < 8; ++r) {
      float cm = fmaxf(fmaxf(s[0].x[r], s[1].x[r]), fmaxf(s[2].x[r], s[3].x[r]));
      cm = fmaxf(cm, __shfl_xor(cm, 1, 32));
      cm = fmaxf(cm, __shfl_xor(cm, 2, 32));
      cm = fmaxf(cm, __shfl_xor(cm, 4, 32));
      cm = fmaxf(cm, __shfl_xor(cm, 8, 32));
      mnew[r] = fmaxf(mrow[r], cm);
      alpha[r] = __expf(mrow[r] - mnew[r]);
      mrow[r] = mnew[r];
    }
#pragma unroll
    for (int kt = 0; kt < 4; ++kt)
#pragma unroll
      for (int r = 0; r < 8; ++r)
        s[kt].x[r] = __expf(s[kt].x[r] - mnew[r]);
#pragma unroll
    for (int r = 0; r < 8; ++r) {
      float rs = s[0].x[r] + s[1].x[r] + s[2].x[r] + s[3].x[r];
      rs += __shfl_xor(rs, 1, 32);
      rs += __shfl_xor(rs, 2, 32);
      rs += __shfl_xor(rs, 4, 32);
      rs += __shfl_xor(rs, 8, 32);
      lrow[r] = lrow[r] * alpha[r] + rs;
    }
#pragma unroll
    for (int t = 0; t < 4; ++t)
#pragma unroll
      for (int r = 0; r < 8; ++r) o[t].x[r] *= alpha[r];
    // P (D-fragment layout) -> LDS -> A-fragment layout
#pragma unroll
    for (int kt = 0; kt < 4; ++kt)
#pragma unroll
      for (int r = 0; r < 8; ++r)
        p_l[(r + hi*8) * 64 + kt*16 + ln] = (h16)s[kt].x[r];
    wave_lds_sync();
    Frag16 ap[2];
    load_frag<true>(ap[0], p_l, ln, 64, 0, hi);
    load_frag<true>(ap[1], p_l, ln, 64, 32, hi);
    wave_lds_sync();
    // O += P V
#pragma unroll
    for (int t = 0; t < 4; ++t) {
      wmma_f16(o[t], ap[0], bv[t*2+0]);
      wmma_f16(o[t], ap[1], bv[t*2+1]);
    }
  }
  h16* ob = omap + (size_t)b * NPIX * C2D;
#pragma unroll
  for (int r = 0; r < 8; ++r) {
    float inv = 1.0f / lrow[r];
    int rr = n0 + r + hi*8;
#pragma unroll
    for (int t = 0; t < 4; ++t)
      ob[(size_t)rr * C2D + t*16 + ln] = (h16)(o[t].x[r] * inv);
  }
}

// --------------------- fused o-conv + up-conv + bias + gamma*y  (K=64 GEMM x 512)
__global__ __launch_bounds__(256) void final_kernel(
    const h16* __restrict__ omap, const h16* __restrict__ mf,
    const float* __restrict__ upb, const float* __restrict__ gamma_p,
    const float* __restrict__ iny, float* __restrict__ outp) {
  const int lane = threadIdx.x & 31;
  const int hi = lane >> 4, ln = lane & 15;
  const int wv = threadIdx.x >> 5;
  const int tile = blockIdx.x;                     // 0..2303
  const int b = tile / 576;
  const int n0 = (tile % 576) * 16;
  const h16* ab = omap + (size_t)b * NPIX * C2D;

  Frag16 a[2];
  load_frag<true>(a[0], ab, n0 + ln, C2D, 0, hi);
  load_frag<true>(a[1], ab, n0 + ln, C2D, 32, hi);

  Frag16 bw[8];
#pragma unroll
  for (int t = 0; t < 4; ++t) {
    load_frag<false>(bw[t*2+0], mf, (wv*4 + t)*16 + ln, C2D, 0,  hi);
    load_frag<false>(bw[t*2+1], mf, (wv*4 + t)*16 + ln, C2D, 32, hi);
  }
  Frag32 acc[4];
#pragma unroll
  for (int t = 0; t < 4; ++t) {
#pragma unroll
    for (int r = 0; r < 8; ++r) acc[t].x[r] = 0.f;
    wmma_f16(acc[t], a[0], bw[t*2+0]);
    wmma_f16(acc[t], a[1], bw[t*2+1]);
  }
  const float gm = *gamma_p;
  const float* yb = iny + (size_t)b * CHN * HIN * HIN;
  float* ob = outp + (size_t)b * CHN * HIN * HIN;
#pragma unroll
  for (int t = 0; t < 4; ++t) {
    int col = (wv*4 + t)*16 + ln;
    int oc = col >> 2, p = (col >> 1) & 1, q = col & 1;
    float bs = upb[oc];
#pragma unroll
    for (int r = 0; r < 8; ++r) {
      int n = n0 + r + hi*8;
      int i = n / HWD, j = n - i * HWD;
      size_t idx = ((size_t)oc * HIN + (2*i + p)) * HIN + (2*j + q);
      ob[idx] = acc[t].x[r] + bs + gm * yb[idx];
    }
  }
}

// ------------------------------------------------------------------- launcher
extern "C" void kernel_launch(void* const* d_in, const int* in_sizes, int n_in,
                              void* d_out, int out_size, void* d_ws, size_t ws_size,
                              hipStream_t stream) {
  (void)in_sizes; (void)n_in; (void)out_size; (void)ws_size;
  const float* input_x = (const float*)d_in[0];
  const float* input_y = (const float*)d_in[1];
  const float* down0_w = (const float*)d_in[2];
  const float* down0_b = (const float*)d_in[3];
  const float* down1_w = (const float*)d_in[4];
  const float* down1_b = (const float*)d_in[5];
  const float* theta_w = (const float*)d_in[6];
  const float* phi_w   = (const float*)d_in[7];
  const float* g_w     = (const float*)d_in[8];
  const float* o_w     = (const float*)d_in[9];
  const float* up_w    = (const float*)d_in[10];
  const float* up_b    = (const float*)d_in[11];
  const float* gamma   = (const float*)d_in[12];
  float* out = (float*)d_out;

  char* ws = (char*)d_ws;
  constexpr size_t SZ_XD = (size_t)NB * NPIX * CHN * 2;   // 9,437,184
  constexpr size_t SZ_T  = (size_t)NB * NPIX * C2D * 2;   // 4,718,592
  constexpr size_t SZ_P  = (size_t)NB * MPIX * C2D * 2;   // 1,179,648
  size_t off = 0;
  h16* xdh     = (h16*)(ws + off); off += SZ_XD;
  h16* ydh     = (h16*)(ws + off); off += SZ_XD;
  h16* theta_h = (h16*)(ws + off); off += SZ_T;
  h16* phifull = (h16*)(ws + off); off += SZ_T;
  h16* gfull   = (h16*)(ws + off); off += SZ_T;
  h16* phi_h   = (h16*)(ws + off); off += SZ_P;
  h16* g_t     = (h16*)(ws + off); off += SZ_P;
  h16* omap_h  = (h16*)(ws + off); off += SZ_T;
  h16* wh0     = (h16*)(ws + off); off += (size_t)CHN*CHN*4*2;
  h16* wh1     = (h16*)(ws + off); off += (size_t)CHN*CHN*4*2;
  h16* thw     = (h16*)(ws + off); off += (size_t)C2D*CHN*2;
  h16* phw     = (h16*)(ws + off); off += (size_t)C2D*CHN*2;
  h16* ghw     = (h16*)(ws + off); off += (size_t)C2D*CHN*2;
  h16* mf      = (h16*)(ws + off); off += (size_t)512*C2D*2;

  prep_convert_kernel<<<64, 256, 0, stream>>>(down0_w, down1_w, theta_w, phi_w, g_w,
                                              wh0, wh1, thw, phw, ghw);
  prep_mfused_kernel<<<128, 256, 0, stream>>>(up_w, o_w, mf);
  down_kernel<<<dim3(288, 2), 256, 0, stream>>>(input_x, input_y, wh0, wh1,
                                                down0_b, down1_b, xdh, ydh);
  proj_kernel<<<dim3(288, 2), 256, 0, stream>>>(xdh, ydh, thw, phw, ghw,
                                                theta_h, phifull, gfull);
  pool_kernel<<<2304, 256, 0, stream>>>(phifull, gfull, phi_h, g_t);
  attn_kernel<<<576, 128, 0, stream>>>(theta_h, phi_h, g_t, omap_h);
  final_kernel<<<2304, 256, 0, stream>>>(omap_h, mf, up_b, gamma, input_y, out);
}
